// MultiHeadAttentionModel_30477087932694
// MI455X (gfx1250) — matrix operand order repaired and checked by reading, not compiled, not run on hardware
//
#include <hip/hip_runtime.h>

// ---------------------------------------------------------------------------
// GPT forward (4L, 8H, C=512, HS=64, V=32000, B=2, T=2048) for gfx1250.
// f16 WMMA everywhere; weights converted+transposed to f16 [N][K] once per
// launch; GEMM tiles streamed into double-buffered LDS by the Tensor Data
// Mover (TDM), pipelined against the WMMAs via in-order TENSORcnt waits.
// Residual stream stays f32.
// ---------------------------------------------------------------------------

typedef _Float16 v16h __attribute__((ext_vector_type(16)));
typedef _Float16 v8h  __attribute__((ext_vector_type(8)));
typedef float    v8f  __attribute__((ext_vector_type(8)));
typedef unsigned int u32x4 __attribute__((ext_vector_type(4)));
typedef int          i32x4 __attribute__((ext_vector_type(4)));
typedef int          i32x8 __attribute__((ext_vector_type(8)));

#define NL 4
#define NH 8
#define CD 512
#define HSZ 64
#define NV 32000
#define NB 2
#define NT 2048
#define MTOT (NB * NT) /* 4096 rows */

#if __has_builtin(__builtin_amdgcn_tensor_load_to_lds)
#define USE_TDM 1
#else
#define USE_TDM 0
#endif

#if USE_TDM
// 2D tile load via TDM: tile0 f16 elements along dim0, tile1 rows, global row
// stride `ld` elements. Descriptor per CDNA5 ISA 8.3/8.4 (D# groups).
__device__ __forceinline__ void tdm_load_2d(const _Float16* gsrc,
                                            unsigned ldsOff, unsigned tile0,
                                            unsigned tile1, unsigned ld,
                                            unsigned rowsAvail) {
  unsigned long long ga = (unsigned long long)gsrc;
  u32x4 g0;
  g0[0] = 1u;                            // count=1 (valid), user mode
  g0[1] = ldsOff;                        // lds_addr (bytes)
  g0[2] = (unsigned)ga;                  // global_addr[31:0]
  g0[3] = ((unsigned)(ga >> 32) & 0x1FFFFFFu) | (2u << 30);  // [56:32]+type=2
  i32x8 g1;
  g1[0] = 1 << 16;                       // data_size=1 (2 bytes), no flags
  g1[1] = (int)((ld & 0xFFFFu) << 16);   // tensor_dim0[15:0] (= ld, no clip)
  g1[2] = (int)(((ld >> 16) & 0xFFFFu) | ((rowsAvail & 0xFFFFu) << 16));
  g1[3] = (int)(((rowsAvail >> 16) & 0xFFFFu) | (tile0 << 16));  // tile_dim0
  g1[4] = (int)tile1;                    // tile_dim1 (tile_dim2 = 0)
  g1[5] = (int)ld;                       // tensor_dim0_stride lo32
  g1[6] = 0;                             // stride hi16 / dim1_stride lo16
  g1[7] = 0;                             // dim1_stride hi32
  i32x4 g2; g2[0] = 1; g2[1] = 1; g2[2] = 0; g2[3] = 0;  // unused upper dims
  i32x4 g3; g3[0] = 0; g3[1] = 0; g3[2] = 0; g3[3] = 0;
#if defined(__clang_major__) && __clang_major__ >= 23
  i32x8 gpad = {0, 0, 0, 0, 0, 0, 0, 0};
  __builtin_amdgcn_tensor_load_to_lds(g0, g1, g2, g3, gpad, 0);
#else
  __builtin_amdgcn_tensor_load_to_lds(g0, g1, g2, g3, 0);
#endif
}
#endif

// ---------------------------- embedding ------------------------------------
__global__ __launch_bounds__(256) void embed_kernel(
    const int* __restrict__ idx, const float* __restrict__ tok,
    const float* __restrict__ pos, float* __restrict__ x) {
  long i = (long)blockIdx.x * 256 + threadIdx.x;  // over MTOT*CD
  int c = (int)(i & (CD - 1));
  long bt = i >> 9;              // CD = 2^9
  int t = (int)(bt & (NT - 1));  // NT = 2^11
  x[i] = tok[(long)idx[bt] * CD + c] + pos[(long)t * CD + c];
}

// -------------------- weight convert + transpose to [N][K] -----------------
// out[n*K + k] = (f16) in[k*sK + (n/64)*sH + (n%64)*sN];  K = 1<<kbits.
__global__ __launch_bounds__(256) void wconv_kernel(
    const float* __restrict__ in, _Float16* __restrict__ out, int kbits,
    long sK, long sN, long sH) {
  long i = (long)blockIdx.x * 256 + threadIdx.x;  // over N*K
  int k = (int)(i & ((1 << kbits) - 1));
  long n = i >> kbits;
  out[i] = (_Float16)in[(long)k * sK + (n >> 6) * sH + (n & 63) * sN];
}

// ------------------------------ f32 -> f16 ---------------------------------
__global__ __launch_bounds__(256) void cvt16_kernel(const float* __restrict__ in,
                                                    _Float16* __restrict__ out) {
  long i = (long)blockIdx.x * 256 + threadIdx.x;
  out[i] = (_Float16)in[i];
}

// ---------------------------- layernorm ------------------------------------
// addto==nullptr: out_f16 = LN(in). addto!=nullptr: addto_f32 += LN(in).
__global__ __launch_bounds__(256) void layernorm_kernel(
    const float* __restrict__ in, const float* __restrict__ g,
    const float* __restrict__ bb, _Float16* __restrict__ out,
    float* __restrict__ addto) {
  __shared__ float red[256];
  int tid = threadIdx.x;
  long row = blockIdx.x;
  const float* xr = in + row * CD;
  float a0 = xr[tid], a1 = xr[tid + 256];
  red[tid] = a0 + a1;
  __syncthreads();
#pragma unroll
  for (int d = 128; d > 0; d >>= 1) {
    if (tid < d) red[tid] += red[tid + d];
    __syncthreads();
  }
  float mean = red[0] * (1.0f / CD);
  __syncthreads();
  float d0 = a0 - mean, d1 = a1 - mean;
  red[tid] = d0 * d0 + d1 * d1;
  __syncthreads();
#pragma unroll
  for (int d = 128; d > 0; d >>= 1) {
    if (tid < d) red[tid] += red[tid + d];
    __syncthreads();
  }
  float rstd = rsqrtf(red[0] * (1.0f / CD) + 1e-5f);
  float v0 = d0 * rstd * g[tid] + bb[tid];
  float v1 = d1 * rstd * g[tid + 256] + bb[tid + 256];
  if (addto) {
    addto[row * CD + tid] += v0;
    addto[row * CD + tid + 256] += v1;
  } else {
    out[row * CD + tid] = (_Float16)v0;
    out[row * CD + tid + 256] = (_Float16)v1;
  }
}

// ------------------------------ WMMA GEMM ----------------------------------
// Out[m,n] = sum_k A[m,k]*Bt[n,k] (+bias) (+residual f32) (relu?).
// A: f16 [M][K]; Bt: f16 [N][K] (pre-transposed). Writes f32 (OutF) or f16
// (OutH). Tile BM=128, BN=64, BK=32; 8 waves, 2x2 WMMA tiles per wave.
// Double-buffered LDS; TDM pipelined one tile ahead (TENSORcnt is in-order:
// after issuing the next pair, wait<=2 guarantees the current pair landed).
#define BM 128
#define BN 64
#define BK 32

__global__ __launch_bounds__(256) void gemm_kernel(
    const _Float16* __restrict__ A, const _Float16* __restrict__ Bt,
    const float* __restrict__ bias, const float* __restrict__ residual,
    float* __restrict__ OutF, _Float16* __restrict__ OutH, int M, int N, int K,
    int relu) {
  __shared__ __align__(16) _Float16 As[2][BM][BK];  // row-major, packed
  __shared__ __align__(16) _Float16 Bs[2][BN][BK];  // [n][k], packed

  int tid = threadIdx.x;
  int m0 = blockIdx.x * BM;
  int n0 = blockIdx.y * BN;
  int lane = tid & 31;
  int w = tid >> 5;
  int wm = w >> 1, wn = w & 1;
  int l16 = lane & 15, lh = lane >> 4;
  int ntiles = K / BK;

  const _Float16* Ab = A + (long)m0 * K;
  const _Float16* Bb = Bt + (long)n0 * K;

  v8f acc[2][2] = {};

#if USE_TDM
  if (tid < 32) {  // wave 0 drives the Tensor Data Mover; preload tile 0
    tdm_load_2d(Ab, (unsigned)(unsigned long long)&As[0][0][0], BK, BM,
                (unsigned)K, (unsigned)(M - m0));
    tdm_load_2d(Bb, (unsigned)(unsigned long long)&Bs[0][0][0], BK, BN,
                (unsigned)K, (unsigned)(N - n0));
  }
#endif

  for (int t = 0; t < ntiles; ++t) {
    int cur = t & 1;
#if USE_TDM
    if (tid < 32) {
      if (t + 1 < ntiles) {  // prefetch next tile into the other buffer
        int nxt = cur ^ 1;
        tdm_load_2d(Ab + (t + 1) * BK,
                    (unsigned)(unsigned long long)&As[nxt][0][0], BK, BM,
                    (unsigned)K, (unsigned)(M - m0));
        tdm_load_2d(Bb + (t + 1) * BK,
                    (unsigned)(unsigned long long)&Bs[nxt][0][0], BK, BN,
                    (unsigned)K, (unsigned)(N - n0));
        __builtin_amdgcn_s_wait_tensorcnt(2);  // current pair has landed
      } else {
        __builtin_amdgcn_s_wait_tensorcnt(0);
      }
    }
#else
    {  // manual vector copies (16B per v8h)
      int k0 = t * BK;
      int r = tid >> 1;
      int c0 = (tid & 1) * 16;
      const _Float16* ap = Ab + (long)r * K + k0 + c0;
      if (k0 + BK < K) __builtin_prefetch(ap + BK, 0, 1);
      *(v8h*)&As[cur][r][c0] = *(const v8h*)ap;
      *(v8h*)&As[cur][r][c0 + 8] = *(const v8h*)(ap + 8);
      int n = tid & 63;
      int kk0 = (tid >> 6) * 8;
      *(v8h*)&Bs[cur][n][kk0] = *(const v8h*)(Bb + (long)n * K + k0 + kk0);
    }
#endif
    __syncthreads();

    v16h af[2], bf[2];
    int b8 = lh * 8;
#pragma unroll
    for (int mt = 0; mt < 2; ++mt) {
      int row = wm * 32 + mt * 16 + l16;
      v8h lo = *(const v8h*)&As[cur][row][b8];       // K = b8..b8+7
      v8h hi = *(const v8h*)&As[cur][row][b8 + 16];  // K = b8+16..b8+23
      af[mt] = __builtin_shufflevector(lo, hi, 0, 1, 2, 3, 4, 5, 6, 7, 8, 9,
                                       10, 11, 12, 13, 14, 15);
    }
    int kb = lh * 8;
#pragma unroll
    for (int nt = 0; nt < 2; ++nt) {
      int col = wn * 32 + nt * 16 + l16;
      v8h lo = *(const v8h*)&Bs[cur][col][kb];      // K = kb..kb+7
      v8h hi = *(const v8h*)&Bs[cur][col][kb + 16];  // K = kb+16..kb+23
      bf[nt] = __builtin_shufflevector(lo, hi, 0, 1, 2, 3, 4, 5, 6, 7, 8, 9,
                                       10, 11, 12, 13, 14, 15);
    }
#pragma unroll
    for (int mt = 0; mt < 2; ++mt)
#pragma unroll
      for (int nt = 0; nt < 2; ++nt)
        acc[mt][nt] = __builtin_amdgcn_wmma_f32_16x16x32_f16(
            false, af[mt], false, bf[nt], (short)0, acc[mt][nt], false, false);
    __syncthreads();
  }

  // Epilogue: C/D layout row = r + 8*lh, col = lane&15.
#pragma unroll
  for (int mt = 0; mt < 2; ++mt)
#pragma unroll
    for (int nt = 0; nt < 2; ++nt) {
      int nn = n0 + wn * 32 + nt * 16 + l16;
      float bv = bias ? bias[nn] : 0.0f;
#pragma unroll
      for (int r = 0; r < 8; ++r) {
        int mm = m0 + wm * 32 + mt * 16 + r + 8 * lh;
        float vv = acc[mt][nt][r] + bv;
        if (residual) vv += residual[(long)mm * N + nn];
        if (relu) vv = fmaxf(vv, 0.0f);
        if (OutH) OutH[(long)mm * N + nn] = (_Float16)vv;
        else OutF[(long)mm * N + nn] = vv;
      }
    }
}

// --------------------------- flash attention -------------------------------
// q,k,v,att: f16, layout [B, T, H*HS] (row stride CD). No 1/sqrt(HS) scale
// (matches reference). 4 waves/block; wave w: 16 query rows.
__global__ __launch_bounds__(128) void attn_kernel(
    const _Float16* __restrict__ q, const _Float16* __restrict__ k,
    const _Float16* __restrict__ v, _Float16* __restrict__ att) {
  __shared__ __align__(16) _Float16 Pbuf[4][16][32];  // per-wave P staging

  int lane = threadIdx.x & 31;
  int w = threadIdx.x >> 5;
  int l16 = lane & 15, lh = lane >> 4;
  int bh = blockIdx.y;
  int b = bh >> 3, h = bh & 7;
  int q0 = blockIdx.x * 64 + w * 16;

  const _Float16* qbase = q + ((long)(b * NT + q0)) * CD + h * HSZ;
  const _Float16* kbase = k + ((long)(b * NT)) * CD + h * HSZ;
  const _Float16* vbase = v + ((long)(b * NT)) * CD + h * HSZ;

  // Q fragments (A layout, 16x32): slice s covers hs-dim K = s*32..s*32+31.
  v16h qf[2];
#pragma unroll
  for (int s = 0; s < 2; ++s) {
    const _Float16* bp = qbase + (long)l16 * CD + s * 32 + lh * 8;
    v8h lo = *(const v8h*)bp;         // K = s*32 + lh*8 + i
    v8h hi = *(const v8h*)(bp + 16);  // K = s*32 + lh*8 + 16 + i
    qf[s] = __builtin_shufflevector(lo, hi, 0, 1, 2, 3, 4, 5, 6, 7, 8, 9, 10,
                                    11, 12, 13, 14, 15);
  }

  float mrow[8], lrow[8];
  v8f o[4] = {};
#pragma unroll
  for (int r = 0; r < 8; ++r) { mrow[r] = -3e38f; lrow[r] = 0.0f; }

  int kmax = q0 + 15;
  for (int j0 = 0; j0 <= kmax; j0 += 32) {
    // ---- S = Q K^T for 32 keys (two 16x16 C tiles) ----
    v8f st[2];
#pragma unroll
    for (int tt = 0; tt < 2; ++tt) {
      int kg = j0 + tt * 16 + l16;  // this lane's key column
      v8f sacc = {};
#pragma unroll
      for (int s = 0; s < 2; ++s) {
        const _Float16* kp = kbase + (long)kg * CD + s * 32 + lh * 8;
        v8h lo = *(const v8h*)kp;  // K(hs) = s*32 + lh*8 + i
        v8h hi = *(const v8h*)(kp + 16);
        v16h bf = __builtin_shufflevector(lo, hi, 0, 1, 2, 3, 4, 5, 6, 7, 8,
                                          9, 10, 11, 12, 13, 14, 15);
        sacc = __builtin_amdgcn_wmma_f32_16x16x32_f16(
            false, qf[s], false, bf, (short)0, sacc, false, false);
      }
#pragma unroll
      for (int r = 0; r < 8; ++r) {
        int qg = q0 + r + 8 * lh;
        if (kg > qg) sacc[r] = -3e38f;  // causal mask
      }
      st[tt] = sacc;
    }

    // ---- online softmax (row reductions across 16-lane column group) ----
#pragma unroll
    for (int r = 0; r < 8; ++r) {
      float mx = fmaxf(st[0][r], st[1][r]);
#pragma unroll
      for (int d = 8; d >= 1; d >>= 1) mx = fmaxf(mx, __shfl_xor(mx, d, 32));
      float mnew = fmaxf(mrow[r], mx);
      float p0 = __expf(st[0][r] - mnew);
      float p1 = __expf(st[1][r] - mnew);
      st[0][r] = p0;
      st[1][r] = p1;
      float sum = p0 + p1;
#pragma unroll
      for (int d = 8; d >= 1; d >>= 1) sum += __shfl_xor(sum, d, 32);
      float alpha = __expf(mrow[r] - mnew);
      lrow[r] = lrow[r] * alpha + sum;
      mrow[r] = mnew;
#pragma unroll
      for (int nt = 0; nt < 4; ++nt) o[nt][r] *= alpha;
    }

    // ---- reshape P (C layout -> A fragment) through per-wave LDS ----
#pragma unroll
    for (int tt = 0; tt < 2; ++tt)
#pragma unroll
      for (int r = 0; r < 8; ++r)
        Pbuf[w][r + 8 * lh][tt * 16 + l16] = (_Float16)st[tt][r];

    v16h pf;
    {
      const _Float16* pr = &Pbuf[w][l16][lh * 8];
      v8h lo = *(const v8h*)pr;         // K(key) = lh*8 + i
      v8h hi = *(const v8h*)(pr + 16);  // K(key) = lh*8 + 16 + i
      pf = __builtin_shufflevector(lo, hi, 0, 1, 2, 3, 4, 5, 6, 7, 8, 9, 10,
                                   11, 12, 13, 14, 15);
    }

    // ---- O += P (16x32) x V (32x64): 4 WMMA over hs column tiles ----
#pragma unroll
    for (int nt = 0; nt < 4; ++nt) {
      const _Float16* vb = vbase + (long)(j0 + lh * 8) * CD + nt * 16 + l16;
      v16h vf;  // B fragment: col = hs, K = key = lh*8 + i (+8 for i >= 8)
#pragma unroll
      for (int i = 0; i < 16; ++i) vf[i] = vb[(long)(i + (i >= 8) * 8) * CD];
      o[nt] = __builtin_amdgcn_wmma_f32_16x16x32_f16(
          false, pf, false, vf, (short)0, o[nt], false, false);
    }
  }

  // ---- normalize and store att[b, qg, h*64 + col] (f16) ----
#pragma unroll
  for (int r = 0; r < 8; ++r) {
    int qg = q0 + r + 8 * lh;
    float inv = 1.0f / lrow[r];
    _Float16* op = att + ((long)(b * NT + qg)) * CD + h * HSZ + l16;
#pragma unroll
    for (int nt = 0; nt < 4; ++nt) op[nt * 16] = (_Float16)(o[nt][r] * inv);
  }
}

// ------------------------------ launcher -----------------------------------
extern "C" void kernel_launch(void* const* d_in, const int* in_sizes, int n_in,
                              void* d_out, int out_size, void* d_ws,
                              size_t ws_size, hipStream_t stream) {
  (void)in_sizes; (void)n_in; (void)out_size; (void)ws_size;
  const int*   idx     = (const int*)d_in[0];
  const float* tok_emb = (const float*)d_in[1];
  const float* pos_emb = (const float*)d_in[2];
  const float* ln1_g   = (const float*)d_in[3];
  const float* ln1_b   = (const float*)d_in[4];
  const float* wq      = (const float*)d_in[5];
  const float* wk      = (const float*)d_in[6];
  const float* wv      = (const float*)d_in[7];
  const float* proj_w  = (const float*)d_in[8];
  const float* proj_b  = (const float*)d_in[9];
  const float* ln2_g   = (const float*)d_in[10];
  const float* ln2_b   = (const float*)d_in[11];
  const float* ff_w1   = (const float*)d_in[12];
  const float* ff_b1   = (const float*)d_in[13];
  const float* ff_w2   = (const float*)d_in[14];
  const float* ff_b2   = (const float*)d_in[15];
  const float* ffln_g  = (const float*)d_in[16];
  const float* ffln_b  = (const float*)d_in[17];
  const float* lm_w    = (const float*)d_in[18];
  const float* lm_b    = (const float*)d_in[19];
  float* out = (float*)d_out;

  // ---- workspace carve (~117 MB); every buffer fully written before read ---
  char* p = (char*)d_ws;
  float* x = (float*)p;          p += (long)MTOT * CD * 4;
  float* fft = (float*)p;        p += (long)MTOT * CD * 4;
  _Float16* hbuf = (_Float16*)p; p += (long)MTOT * CD * 2;
  _Float16* qh = (_Float16*)p;   p += (long)MTOT * CD * 2;
  _Float16* kh = (_Float16*)p;   p += (long)MTOT * CD * 2;
  _Float16* vh = (_Float16*)p;   p += (long)MTOT * CD * 2;
  _Float16* atth = (_Float16*)p; p += (long)MTOT * CD * 2;
  _Float16* ff1h = (_Float16*)p; p += (long)MTOT * 4 * CD * 2;
  _Float16* xh = (_Float16*)p;   p += (long)MTOT * CD * 2;
  _Float16* wT = (_Float16*)p;   // transposed f16 weights

  const long WQKV = (long)NH * CD * HSZ;  // 262144, per-layer qkv weight
  const long LWT = 3 * 262144 + 262144 + 2 * ((long)CD * 4 * CD);  // 3145728
  dim3 blk256(256), blk128(128);

  // ---- one-time (per launch) weight convert + transpose to f16 [N][K] ----
  for (int l = 0; l < NL; ++l) {
    _Float16* lw = wT + l * LWT;
    // q/k/v: in[k*64 + (n/64)*32768 + (n%64)]  (K=512 -> kbits=9)
    wconv_kernel<<<dim3((CD * CD) / 256), blk256, 0, stream>>>(
        wq + l * WQKV, lw + 0 * 262144, 9, HSZ, 1, (long)CD * HSZ);
    wconv_kernel<<<dim3((CD * CD) / 256), blk256, 0, stream>>>(
        wk + l * WQKV, lw + 1 * 262144, 9, HSZ, 1, (long)CD * HSZ);
    wconv_kernel<<<dim3((CD * CD) / 256), blk256, 0, stream>>>(
        wv + l * WQKV, lw + 2 * 262144, 9, HSZ, 1, (long)CD * HSZ);
    // proj^T: in[n*512 + k]
    wconv_kernel<<<dim3((CD * CD) / 256), blk256, 0, stream>>>(
        proj_w + (long)l * CD * CD, lw + 3 * 262144, 9, 1, CD, (long)64 * CD);
    // ff_w1 row-major [512][2048]
    wconv_kernel<<<dim3((CD * 4 * CD) / 256), blk256, 0, stream>>>(
        ff_w1 + (long)l * CD * 4 * CD, lw + 4 * 262144, 9, 4 * CD, 1, 64);
    // ff_w2 row-major [2048][512] (K=2048 -> kbits=11)
    wconv_kernel<<<dim3((CD * 4 * CD) / 256), blk256, 0, stream>>>(
        ff_w2 + (long)l * 4 * CD * CD, lw + 8 * 262144, 11, CD, 1, 64);
  }
  _Float16* lmT = wT + NL * LWT;  // [V][C]
  wconv_kernel<<<dim3(((long)NV * CD) / 256), blk256, 0, stream>>>(
      lm_w, lmT, 9, NV, 1, 64);

  embed_kernel<<<dim3((MTOT * CD) / 256), blk256, 0, stream>>>(idx, tok_emb,
                                                               pos_emb, x);

  dim3 gCC(MTOT / BM, CD / BN);  // (32, 8)
  for (int l = 0; l < NL; ++l) {
    _Float16* lw = wT + l * LWT;
    layernorm_kernel<<<dim3(MTOT), blk256, 0, stream>>>(
        x, ln1_g + l * CD, ln1_b + l * CD, hbuf, nullptr);
    gemm_kernel<<<gCC, blk256, 0, stream>>>(hbuf, lw + 0 * 262144, nullptr,
                                            nullptr, nullptr, qh, MTOT, CD,
                                            CD, 0);
    gemm_kernel<<<gCC, blk256, 0, stream>>>(hbuf, lw + 1 * 262144, nullptr,
                                            nullptr, nullptr, kh, MTOT, CD,
                                            CD, 0);
    gemm_kernel<<<gCC, blk256, 0, stream>>>(hbuf, lw + 2 * 262144, nullptr,
                                            nullptr, nullptr, vh, MTOT, CD,
                                            CD, 0);
    attn_kernel<<<dim3(NT / 64, NB * NH), blk128, 0, stream>>>(qh, kh, vh,
                                                               atth);
    // x = x + att @ proj^T + proj_b
    gemm_kernel<<<gCC, blk256, 0, stream>>>(atth, lw + 3 * 262144,
                                            proj_b + l * CD, x, x, nullptr,
                                            MTOT, CD, CD, 0);
    layernorm_kernel<<<dim3(MTOT), blk256, 0, stream>>>(
        x, ln2_g + l * CD, ln2_b + l * CD, hbuf, nullptr);
    // ff1 = relu(h @ ff_w1 + b1) -> f16
    gemm_kernel<<<dim3(MTOT / BM, (4 * CD) / BN), blk256, 0, stream>>>(
        hbuf, lw + 4 * 262144, ff_b1 + (long)l * 4 * CD, nullptr, nullptr,
        ff1h, MTOT, 4 * CD, CD, 1);
    // fft = ff1 @ ff_w2 + b2 -> f32
    gemm_kernel<<<gCC, blk256, 0, stream>>>(ff1h, lw + 8 * 262144,
                                            ff_b2 + l * CD, nullptr, fft,
                                            nullptr, MTOT, CD, 4 * CD, 0);
    // x = x + LN(fft)
    layernorm_kernel<<<dim3(MTOT), blk256, 0, stream>>>(
        fft, ffln_g + l * CD, ffln_b + l * CD, nullptr, x);
  }

  // logits = x @ lm_w + lm_b  (A = f16 copy of x)
  cvt16_kernel<<<dim3((MTOT * CD) / 256), blk256, 0, stream>>>(x, xh);
  gemm_kernel<<<dim3(MTOT / BM, NV / BN), blk256, 0, stream>>>(
      xh, lmT, lm_b, nullptr, out, nullptr, MTOT, NV, CD, 0);
}
